// FineGrainedOpLSTMCell_68968584839425
// MI455X (gfx1250) — matrix operand order repaired and checked by reading, not compiled, not run on hardware
//
#include <hip/hip_runtime.h>
#include <math.h>

// ---------------------------------------------------------------------------
// LSTM cell on MI455X (gfx1250): bf16 WMMA GEMMs + fused gate epilogue.
//   B=4096, INPUT=HIDDEN=1024.
//   Workspace layout (needs 32 MB):
//     [0,        8MB)  : packed A fragments of x      (bf16)
//     [8MB,     16MB)  : packed A fragments of h_prev (bf16)
//     [16MB,    32MB)  : packed B fragments of Wi,Wf,Wg,Wo,Ui,Uf,Ug,Uo (bf16)
// ---------------------------------------------------------------------------

typedef __attribute__((ext_vector_type(16))) __bf16 v16bf;
typedef __attribute__((ext_vector_type(2)))  __bf16 v2bf;
typedef __attribute__((ext_vector_type(8)))  float  v8f;

#define BDIM   4096
#define KDIM   1024
#define NDIM   1024
#define KSTEPS (KDIM / 32)   // 32
#define MTILES (BDIM / 16)   // 256
#define NTILES (NDIM / 16)   // 64
// v16bf elements per packed weight matrix (= 2 MB / 32 B)
#define BFRAGS (NTILES * KSTEPS * 32)   // 65536

// ---------------------------------------------------------------------------
// Pack A (x or h_prev): f32 row-major [4096 x 1024] -> bf16 WMMA A fragments.
// 16-bit A 16x32 layout (ISA 7.12.2): lane l holds row M=l%16; VGPR pair v:
//   K = (v<4 ? 0 : 16) + (l/16)*8 + 2*(v&3) .. +1
// Fragment stored lane-major: dword index = frag*256 + lane*8 + v.
// One thread per packed dword (2 bf16).
// ---------------------------------------------------------------------------
__global__ __launch_bounds__(256)
void pack_a_kernel(const float* __restrict__ src, v2bf* __restrict__ dst) {
    int blk   = blockIdx.x;           // = mtile*KSTEPS + kstep
    int t     = threadIdx.x;          // 0..255
    int lane  = t >> 3;               // 0..31
    int v     = t & 7;                // 0..7
    int mtile = blk / KSTEPS;
    int kstep = blk % KSTEPS;
    int g     = lane >> 4;            // lane half
    int m     = lane & 15;            // row within tile
    int kbase = ((v < 4) ? 0 : 16) + g * 8 + 2 * (v & 3);
    int k     = kstep * 32 + kbase;
    const float* s = src + (size_t)(mtile * 16 + m) * KDIM + k;
    v2bf d;
    d[0] = (__bf16)s[0];
    d[1] = (__bf16)s[1];
    dst[(size_t)blk * 256 + t] = d;
}

// ---------------------------------------------------------------------------
// Pack B (weight matrix): f32 row-major [K=1024 x N=1024] -> bf16 B fragments.
// 16-bit B 32x16 layout: lane l holds column N=l%16; VGPR pair v:
//   K = (v<4 ? 0 : 16) + (l/16)*8 + 2*(v&3) .. +1
// ---------------------------------------------------------------------------
__global__ __launch_bounds__(256)
void pack_b_kernel(const float* __restrict__ src, v2bf* __restrict__ dst) {
    int blk   = blockIdx.x;           // = ntile*KSTEPS + kstep
    int t     = threadIdx.x;
    int lane  = t >> 3;
    int v     = t & 7;
    int ntile = blk / KSTEPS;
    int kstep = blk % KSTEPS;
    int g     = lane >> 4;
    int n     = lane & 15;
    int k     = kstep * 32 + ((v < 4) ? 0 : 16) + g * 8 + 2 * (v & 3);
    const float* s = src + (size_t)k * NDIM + (ntile * 16 + n);
    v2bf d;
    d[0] = (__bf16)s[0];
    d[1] = (__bf16)s[NDIM];           // K+1 row
    dst[(size_t)blk * 256 + t] = d;
}

// ---------------------------------------------------------------------------
// Fused GEMM + LSTM gate kernel.
// Each wave: 2 M-tiles (32 rows) x 1 N-tile x 4 gates -> 8 f32 accumulators.
// Per K-step: 4 A b128-loads + 8 B b128-loads feed 8 WMMAs (1.5 loads/WMMA).
// All 8 waves of a block share the same N-tile => identical B fragment
// stream per wave => WGP-cache/L2 locality for weights.
// Gate order matches packed order Wi,Wf,Wg,Wo / Ui,Uf,Ug,Uo.
// ---------------------------------------------------------------------------
__global__ __launch_bounds__(256)
void lstm_gemm_kernel(const v16bf* __restrict__ Ax,
                      const v16bf* __restrict__ Ah,
                      const v16bf* __restrict__ Bmats,   // 8 packed matrices
                      const float* __restrict__ c_prev,
                      const float* __restrict__ bi,
                      const float* __restrict__ bf_,
                      const float* __restrict__ bg,
                      const float* __restrict__ bo,
                      float* __restrict__ out_h,
                      float* __restrict__ out_c) {
    int wave  = threadIdx.x >> 5;
    int lane  = threadIdx.x & 31;
    // Block mapping: ntile shared by all 8 waves; waves take 2 M-tiles each.
    int ntile  = blockIdx.x & (NTILES - 1);
    int mgroup = blockIdx.x >> 6;            // 0..15, each covers 16 M-tiles
    int mtileA = mgroup * 16 + wave * 2;     // first of the wave's 2 M-tiles

    v8f acc[2][4] = {};

    // Phase 0: x @ {Wi,Wf,Wg,Wo};  Phase 1: h_prev @ {Ui,Uf,Ug,Uo}
    for (int ph = 0; ph < 2; ++ph) {
        const v16bf* Abase = ph ? Ah : Ax;
        const v16bf* Ap0 = Abase + (size_t)mtileA       * KSTEPS * 32 + lane;
        const v16bf* Ap1 = Abase + (size_t)(mtileA + 1) * KSTEPS * 32 + lane;
        const v16bf* Bp  = Bmats + (size_t)(ph * 4) * BFRAGS
                                 + (size_t)ntile * KSTEPS * 32 + lane;
        for (int ks = 0; ks < KSTEPS; ++ks) {
            v16bf a0 = Ap0[ks * 32];
            v16bf a1 = Ap1[ks * 32];
#pragma unroll
            for (int gate = 0; gate < 4; ++gate) {
                v16bf b = Bp[(size_t)gate * BFRAGS + ks * 32];
                acc[0][gate] = __builtin_amdgcn_wmma_f32_16x16x32_bf16(
                    false, a0, false, b, (short)0, acc[0][gate], false, false);
                acc[1][gate] = __builtin_amdgcn_wmma_f32_16x16x32_bf16(
                    false, a1, false, b, (short)0, acc[1][gate], false, false);
            }
        }
    }

    // Epilogue: C/D layout (7.12.2): lane l, VGPR v -> M = v + 8*(l/16),
    // N = l%16.  Fuse bias + activations + cell/hidden update.
    int n    = lane & 15;
    int half = lane >> 4;
    int col  = ntile * 16 + n;
    float bbi = bi[col], bbf = bf_[col], bbg = bg[col], bbo = bo[col];
#pragma unroll
    for (int rt = 0; rt < 2; ++rt) {
        int rowBase = (mtileA + rt) * 16 + half * 8;
#pragma unroll
        for (int v = 0; v < 8; ++v) {
            int r = rowBase + v;
            size_t idx = (size_t)r * NDIM + col;
            float zi = acc[rt][0][v] + bbi;
            float zf = acc[rt][1][v] + bbf;
            float zg = acc[rt][2][v] + bbg;
            float zo = acc[rt][3][v] + bbo;
            float ig = 1.0f / (1.0f + __expf(-zi));
            float fg = 1.0f / (1.0f + __expf(-zf));
            float og = 1.0f / (1.0f + __expf(-zo));
            float gg = tanhf(zg);
            float cell = c_prev[idx] * fg + gg * ig;
            float hid  = tanhf(cell) * og;
            out_h[idx] = hid;
            out_c[idx] = cell;
        }
    }
}

// ---------------------------------------------------------------------------
extern "C" void kernel_launch(void* const* d_in, const int* in_sizes, int n_in,
                              void* d_out, int out_size, void* d_ws, size_t ws_size,
                              hipStream_t stream) {
    const float* x  = (const float*)d_in[0];
    const float* h  = (const float*)d_in[1];
    const float* c  = (const float*)d_in[2];
    const float* Wi = (const float*)d_in[3];
    const float* Ui = (const float*)d_in[4];
    const float* bi = (const float*)d_in[5];
    const float* Wf = (const float*)d_in[6];
    const float* Uf = (const float*)d_in[7];
    const float* bf = (const float*)d_in[8];
    const float* Wg = (const float*)d_in[9];
    const float* Ug = (const float*)d_in[10];
    const float* bg = (const float*)d_in[11];
    const float* Wo = (const float*)d_in[12];
    const float* Uo = (const float*)d_in[13];
    const float* bo = (const float*)d_in[14];

    char* ws = (char*)d_ws;
    const size_t aBytes = (size_t)BDIM * KDIM * 2;   // 8 MB per activation
    const size_t bBytes = (size_t)KDIM * NDIM * 2;   // 2 MB per weight matrix

    v2bf* Ax = (v2bf*)ws;
    v2bf* Ah = (v2bf*)(ws + aBytes);
    char* Bbase = ws + 2 * aBytes;                   // 8 matrices follow

    // Pack activations (x, h_prev) into A fragments.
    pack_a_kernel<<<MTILES * KSTEPS, 256, 0, stream>>>(x, Ax);
    pack_a_kernel<<<MTILES * KSTEPS, 256, 0, stream>>>(h, Ah);

    // Pack weights into B fragments: order Wi,Wf,Wg,Wo,Ui,Uf,Ug,Uo.
    const float* wsrc[8] = {Wi, Wf, Wg, Wo, Ui, Uf, Ug, Uo};
    for (int m = 0; m < 8; ++m) {
        pack_b_kernel<<<NTILES * KSTEPS, 256, 0, stream>>>(
            wsrc[m], (v2bf*)(Bbase + (size_t)m * bBytes));
    }

    // Fused GEMM + gates. Output: hidden then cell, each 4096x1024 f32.
    // Blocks: 16 M-groups x 64 N-tiles; 8 waves/block, 2 M-tiles per wave.
    float* out_h = (float*)d_out;
    float* out_c = out_h + (size_t)BDIM * NDIM;
    lstm_gemm_kernel<<<(MTILES / 16) * NTILES, 256, 0, stream>>>(
        (const v16bf*)Ax, (const v16bf*)Ah, (const v16bf*)Bbase,
        c, bi, bf, bg, bo, out_h, out_c);
}
